// TransformerBlock_17763984736704
// MI455X (gfx1250) — matrix operand order repaired and checked by reading, not compiled, not run on hardware
//
#include <hip/hip_runtime.h>
#include <math.h>

// ---------------------------------------------------------------------------
// Transformer block (B=1, S=2048, DIM=2048, H=16, HK=8, HD=128, HIDDEN=5632)
// CDNA5 / gfx1250: all GEMMs + attention via v_wmma_f32_16x16x32_f16 (wave32),
// LDS staging via GLOBAL_LOAD_ASYNC_TO_LDS_B128 (ASYNCcnt) where possible,
// 2-stage double-buffered pipelines in GEMM and flash attention.
// ---------------------------------------------------------------------------

typedef __attribute__((ext_vector_type(16))) _Float16 v16h;
typedef __attribute__((ext_vector_type(8)))  _Float16 h8;
typedef __attribute__((ext_vector_type(8)))  float    v8f;
typedef int v4i __attribute__((ext_vector_type(4)));
typedef __attribute__((address_space(1))) v4i* as1_v4i_ptr;  // global
typedef __attribute__((address_space(3))) v4i* as3_v4i_ptr;  // LDS

#define S_LEN  2048
#define DIM    2048
#define NH     16
#define NKV    8
#define HD     128
#define ROT    32
#define HIDDEN 5632
#define EPSV   1e-6f

// ---- CDNA5 async global->LDS copy (ASYNCcnt), with portable fallback ------
#if defined(__has_builtin)
#if __has_builtin(__builtin_amdgcn_global_load_async_to_lds_b128)
#define HAVE_ASYNC_LDS 1
#endif
#endif
#ifndef HAVE_ASYNC_LDS
#define HAVE_ASYNC_LDS 0
#endif

#if HAVE_ASYNC_LDS
__device__ __forceinline__ void async_copy16(const _Float16* g, _Float16* l) {
  // signature (probe-derived): (v4i AS1* src, v4i AS3* dst, imm offset, imm cpol)
  as1_v4i_ptr gp = (as1_v4i_ptr)(unsigned long long)(uintptr_t)g;
  as3_v4i_ptr lp = (as3_v4i_ptr)(unsigned int)(uintptr_t)l;
  __builtin_amdgcn_global_load_async_to_lds_b128(gp, lp, 0, 0);
}
__device__ __forceinline__ void async_wait_all() {
#if __has_builtin(__builtin_amdgcn_s_wait_asynccnt)
  __builtin_amdgcn_s_wait_asynccnt(0);
#else
  asm volatile("s_wait_asynccnt 0x0" ::: "memory");
#endif
}
#else
__device__ __forceinline__ void async_copy16(const _Float16* g, _Float16* l) {
  *(h8*)l = *(const h8*)g;   // sync fallback: global load + ds store
}
__device__ __forceinline__ void async_wait_all() {}
#endif

// ---- WMMA fragment loaders (ISA 7.12.2 layouts, wave32) --------------------
// A matrix 16x32 f16: lane L -> row L%16; elems 0..7 = K klo..klo+7,
// elems 8..15 = K 16+klo..; klo = (L>=16)?8:0.
__device__ __forceinline__ v16h make_a_frag(const _Float16* base, int lane) {
  const int klo = (lane >= 16) ? 8 : 0;
  h8 lo = *(const h8*)(base + klo);
  h8 hi = *(const h8*)(base + 16 + klo);
  v16h r;
#pragma unroll
  for (int i = 0; i < 8; ++i) { r[i] = lo[i]; r[i + 8] = hi[i]; }
  return r;
}
// B matrix 32x16 f16: lane L -> col L%16; elems 0..7 = K kh..kh+7,
// elems 8..15 = K 16+kh..; kh = (L>=16)?8:0.  rowbase = &lds[col * ld].
__device__ __forceinline__ v16h make_b_frag(const _Float16* rowbase, int lane) {
  const int kh = (lane >= 16) ? 8 : 0;
  h8 lo = *(const h8*)(rowbase + kh);
  h8 hi = *(const h8*)(rowbase + 16 + kh);
  v16h r;
#pragma unroll
  for (int i = 0; i < 8; ++i) { r[i] = lo[i]; r[i + 8] = hi[i]; }
  return r;
}

__device__ __forceinline__ v8f zero_v8f() {
  v8f z;
#pragma unroll
  for (int i = 0; i < 8; ++i) z[i] = 0.f;
  return z;
}

// ---- RMSNorm (f32 in) -> f16 out, one block per row of 2048 ---------------
__global__ __launch_bounds__(256) void rmsnorm_f16(
    const float* __restrict__ x, const float* __restrict__ w,
    _Float16* __restrict__ out, int n) {
  const int row = blockIdx.x, tid = threadIdx.x;
  const float* xr = x + (size_t)row * n;
  float4 v0 = ((const float4*)xr)[tid];
  float4 v1 = ((const float4*)xr)[tid + 256];
  float a[8] = {v0.x, v0.y, v0.z, v0.w, v1.x, v1.y, v1.z, v1.w};
  float ss = 0.f;
#pragma unroll
  for (int i = 0; i < 8; ++i) ss += a[i] * a[i];
#pragma unroll
  for (int m = 16; m >= 1; m >>= 1) ss += __shfl_xor(ss, m);
  __shared__ float red[8];
  if ((tid & 31) == 0) red[tid >> 5] = ss;
  __syncthreads();
  float tot = 0.f;
#pragma unroll
  for (int i = 0; i < 8; ++i) tot += red[i];
  const float inv = rsqrtf(tot / (float)n + EPSV);
  const int b0 = tid * 4, b1 = (tid + 256) * 4;
  _Float16* orow = out + (size_t)row * n;
#pragma unroll
  for (int k = 0; k < 4; ++k) {
    orow[b0 + k] = (_Float16)(a[k]     * inv * (1.f + w[b0 + k]));
    orow[b1 + k] = (_Float16)(a[4 + k] * inv * (1.f + w[b1 + k]));
  }
}

// ---- per-head RMSNorm + RoPE (first 32 dims) -> f16 -----------------------
__global__ __launch_bounds__(128) void qk_norm_rope(
    const float* __restrict__ in, int rowStride, int headSpan,
    const float* __restrict__ nw, const float* __restrict__ cosT,
    const float* __restrict__ sinT, _Float16* __restrict__ out,
    int outRowStride) {
  const int s = blockIdx.x, h = blockIdx.y, d = threadIdx.x;  // d in 0..127
  float v = in[(size_t)s * rowStride + h * headSpan + d];
  float ss = v * v;
#pragma unroll
  for (int m = 16; m >= 1; m >>= 1) ss += __shfl_xor(ss, m);
  __shared__ float red[4];
  if ((d & 31) == 0) red[d >> 5] = ss;
  __syncthreads();
  const float tot = red[0] + red[1] + red[2] + red[3];
  float y = v * rsqrtf(tot / (float)HD + EPSV) * (1.f + nw[d]);
  __shared__ float sh[ROT];
  if (d < ROT) sh[d] = y;
  __syncthreads();
  float o = y;
  if (d < 16)        o = y * cosT[s * ROT + d] - sh[d + 16] * sinT[s * ROT + d];
  else if (d < ROT)  o = y * cosT[s * ROT + d] + sh[d - 16] * sinT[s * ROT + d];
  out[(size_t)s * outRowStride + h * HD + d] = (_Float16)o;
}

// ---- plain f32 -> f16 convert ---------------------------------------------
__global__ __launch_bounds__(256) void cvt_f32_f16(
    const float* __restrict__ in, _Float16* __restrict__ out, int n) {
  int i = (blockIdx.x * blockDim.x + threadIdx.x) * 4;
  if (i < n) {
    float4 v = *(const float4*)(in + i);
    out[i] = (_Float16)v.x; out[i + 1] = (_Float16)v.y;
    out[i + 2] = (_Float16)v.z; out[i + 3] = (_Float16)v.w;
  }
}

// ---- WMMA GEMM: C[M,N] = A_f16[M,K] @ W_f32[N,K]^T ------------------------
// Double-buffered: A tile via async global->LDS copy; W tile f32 in regs,
// converted to f16 and stored to LDS at the top of the next iteration.
// EPI 0: outF = acc
// EPI 1: outF = aux + acc                        (residual add)
// EPI 2: outH = silu(aux) * acc  (f16 out)       (swiglu fuse)
template <int EPI>
__global__ __launch_bounds__(256) void gemm_f16(
    const _Float16* __restrict__ A, const float* __restrict__ W,
    float* __restrict__ outF, _Float16* __restrict__ outH,
    const float* __restrict__ aux, int M, int N, int K) {
  constexpr int BM = 128, BN = 128, BK = 32, LDA = 40, LDB = 40;
  __shared__ _Float16 As[2][BM * LDA];
  __shared__ _Float16 Bs[2][BN * LDB];
  const int tid = threadIdx.x, lane = tid & 31, wv = tid >> 5;
  const int bm = blockIdx.y * BM, bn = blockIdx.x * BN;
  const int wm = (wv & 3) * 32, wn = (wv >> 2) * 64;
  const int ln = lane & 15, lh = lane >> 4;

  v8f acc[2][4];
#pragma unroll
  for (int i = 0; i < 2; ++i)
#pragma unroll
    for (int j = 0; j < 4; ++j) acc[i][j] = zero_v8f();

  const int ar = tid >> 1, ah = (tid & 1) * 16;  // 128 rows x 2 half-rows
  const _Float16* Ag = A + (size_t)(bm + ar) * K + ah;
  const float*    Wg = W + (size_t)(bn + ar) * K + ah;

  float4 wr0, wr1, wr2, wr3;   // in-flight f32 weight tile (registers)
  auto load_w = [&](int k0) {
    const float4* p = (const float4*)(Wg + k0);
    wr0 = p[0]; wr1 = p[1]; wr2 = p[2]; wr3 = p[3];
  };
  auto issue_a = [&](int buf, int k0) {
    _Float16* l = &As[buf][ar * LDA + ah];
    async_copy16(Ag + k0,     l);
    async_copy16(Ag + k0 + 8, l + 8);
  };
  auto store_w = [&](int buf) {
    h8 p0, p1;
    p0[0]=(_Float16)wr0.x; p0[1]=(_Float16)wr0.y; p0[2]=(_Float16)wr0.z; p0[3]=(_Float16)wr0.w;
    p0[4]=(_Float16)wr1.x; p0[5]=(_Float16)wr1.y; p0[6]=(_Float16)wr1.z; p0[7]=(_Float16)wr1.w;
    p1[0]=(_Float16)wr2.x; p1[1]=(_Float16)wr2.y; p1[2]=(_Float16)wr2.z; p1[3]=(_Float16)wr2.w;
    p1[4]=(_Float16)wr3.x; p1[5]=(_Float16)wr3.y; p1[6]=(_Float16)wr3.z; p1[7]=(_Float16)wr3.w;
    *(h8*)(&Bs[buf][ar * LDB + ah])     = p0;
    *(h8*)(&Bs[buf][ar * LDB + ah + 8]) = p1;
  };

  load_w(0);
  issue_a(0, 0);
  const int nt = K / BK;
  for (int t = 0; t < nt; ++t) {
    const int buf = t & 1;
    store_w(buf);                 // W(t) regs -> LDS
    async_wait_all();             // A(t) async copy landed (this wave)
    __syncthreads();              // all waves' tiles visible
    if (t + 1 < nt) {             // overlap next tile with compute
      load_w((t + 1) * BK);
      issue_a(buf ^ 1, (t + 1) * BK);
    }
    if (t + 2 < nt) __builtin_prefetch(Wg + (t + 2) * BK, 0, 0);

    const _Float16* Ab = As[buf];
    const _Float16* Bb = Bs[buf];
    v16h af[2];
#pragma unroll
    for (int i = 0; i < 2; ++i)
      af[i] = make_a_frag(Ab + (wm + 16 * i + ln) * LDA, lane);
#pragma unroll
    for (int j = 0; j < 4; ++j) {
      v16h bf = make_b_frag(Bb + (wn + 16 * j + ln) * LDB, lane);
#pragma unroll
      for (int i = 0; i < 2; ++i)
        acc[i][j] = __builtin_amdgcn_wmma_f32_16x16x32_f16(
            false, af[i], false, bf, (short)0, acc[i][j], false, false);
    }
  }

  // epilogue: C layout: element r of lane -> (m = r + 8*lh, n = ln)
#pragma unroll
  for (int i = 0; i < 2; ++i)
#pragma unroll
    for (int j = 0; j < 4; ++j)
#pragma unroll
      for (int r = 0; r < 8; ++r) {
        const int gm = bm + wm + 16 * i + r + 8 * lh;
        const int gn = bn + wn + 16 * j + ln;
        const size_t idx = (size_t)gm * N + gn;
        const float v = acc[i][j][r];
        if constexpr (EPI == 0) {
          outF[idx] = v;
        } else if constexpr (EPI == 1) {
          outF[idx] = aux[idx] + v;
        } else {
          const float u = aux[idx];
          outH[idx] = (_Float16)((u / (1.f + __expf(-u))) * v);
        }
      }
}

// ---- Flash attention with WMMA (causal, GQA repeat=2, fused sigmoid gate) --
// Double-buffered K/V tiles: K via async global->LDS, V transposed through
// registers (stored at the top of the next iteration).
__global__ __launch_bounds__(256) void flash_attn(
    const _Float16* __restrict__ Q, const _Float16* __restrict__ Kg,
    const _Float16* __restrict__ Vg, const float* __restrict__ QG,
    _Float16* __restrict__ ctx) {
  constexpr int LDK = 136, LDV = 40, LDP = 40;
  __shared__ _Float16 Kt[2][32 * LDK];     // [key][d]
  __shared__ _Float16 Vt[2][128 * LDV];    // [d][key] (transposed)
  __shared__ _Float16 Pl[8 * 16 * LDP];    // per-wave P transpose scratch
  const int tid = threadIdx.x, lane = tid & 31, wv = tid >> 5;
  const int qb = blockIdx.x, h = blockIdx.y, hk = h >> 1;
  const int qs = qb * 128 + wv * 16;       // this wave's 16 query rows
  const int ln = lane & 15, lh = lane >> 4;

  // Q fragments (16 rows x 128 dims = 4 A-frags), kept resident
  v16h aQ[4];
  const _Float16* qrow = Q + (size_t)(qs + ln) * (NH * HD) + h * HD;
#pragma unroll
  for (int c = 0; c < 4; ++c) aQ[c] = make_a_frag(qrow + c * 32, lane);

  v8f o[8];
  float mrow[8], lrow[8];
#pragma unroll
  for (int t = 0; t < 8; ++t) o[t] = zero_v8f();
#pragma unroll
  for (int r = 0; r < 8; ++r) { mrow[r] = -1e30f; lrow[r] = 0.f; }

  const int nkb = (qb + 1) * 4;            // key blocks of 32 for this block
  const int rmax = qs + 15;                // wave-uniform causal horizon
  _Float16* Pw = Pl + wv * 16 * LDP;
  const float scl = 0.08838834764831845f;  // 1/sqrt(128)

  // cooperative staging roles
  const int key = tid >> 3, seg = tid & 7;
  h8 vr0, vr1;                             // in-flight V tile (registers)
  auto load_v = [&](int kb) {
    const int sk = kb * 32 + key;
    const h8* vs = (const h8*)(Vg + (size_t)sk * (NKV * HD) + hk * HD + seg * 16);
    vr0 = vs[0]; vr1 = vs[1];
  };
  auto issue_k = [&](int buf, int kb) {
    const int sk = kb * 32 + key;
    const _Float16* g = Kg + (size_t)sk * (NKV * HD) + hk * HD + seg * 16;
    _Float16* l = &Kt[buf][key * LDK + seg * 16];
    async_copy16(g,     l);
    async_copy16(g + 8, l + 8);
  };
  auto store_v = [&](int buf) {
#pragma unroll
    for (int e = 0; e < 8; ++e) {
      Vt[buf][(seg * 16 + e)     * LDV + key] = vr0[e];
      Vt[buf][(seg * 16 + 8 + e) * LDV + key] = vr1[e];
    }
  };

  load_v(0);
  issue_k(0, 0);
  for (int kb = 0; kb < nkb; ++kb) {
    const int buf = kb & 1;
    store_v(buf);                 // V(kb) regs -> LDS (transposed)
    async_wait_all();             // K(kb) async copy landed (this wave)
    __syncthreads();
    if (kb + 1 < nkb) {           // overlap next K/V tile with compute
      load_v(kb + 1);
      issue_k(buf ^ 1, kb + 1);
    }

    if (kb * 32 <= rmax) {  // wave-uniform: EXEC stays all-ones for WMMA
      // S = Q K^T : contraction over d (4 chunks of 32), 2 key sub-tiles
      v8f sc[2] = {zero_v8f(), zero_v8f()};
#pragma unroll
      for (int c = 0; c < 4; ++c)
#pragma unroll
        for (int j = 0; j < 2; ++j) {
          v16h bK = make_b_frag(Kt[buf] + (j * 16 + ln) * LDK + c * 32, lane);
          sc[j] = __builtin_amdgcn_wmma_f32_16x16x32_f16(
              false, aQ[c], false, bK, (short)0, sc[j], false, false);
        }

      // scale + causal mask
      float p[2][8];
#pragma unroll
      for (int j = 0; j < 2; ++j)
#pragma unroll
        for (int r = 0; r < 8; ++r) {
          float sv = sc[j][r] * scl;
          const int col = kb * 32 + j * 16 + ln;
          const int row = qs + r + 8 * lh;
          if (col > row) sv = -1e30f;
          p[j][r] = sv;
        }

      // online softmax update (row lives across a 16-lane half-group)
#pragma unroll
      for (int r = 0; r < 8; ++r) {
        float rm = fmaxf(p[0][r], p[1][r]);
#pragma unroll
        for (int m = 8; m >= 1; m >>= 1) rm = fmaxf(rm, __shfl_xor(rm, m));
        const float mn = fmaxf(mrow[r], rm);
        const float al = __expf(mrow[r] - mn);
        const float e0 = __expf(p[0][r] - mn);
        const float e1 = __expf(p[1][r] - mn);
        float rs = e0 + e1;
#pragma unroll
        for (int m = 8; m >= 1; m >>= 1) rs += __shfl_xor(rs, m);
        lrow[r] = lrow[r] * al + rs;
        mrow[r] = mn;
        p[0][r] = e0; p[1][r] = e1;
#pragma unroll
        for (int t = 0; t < 8; ++t) o[t][r] *= al;
      }

      // P: C-layout -> A-layout via per-wave LDS scratch
#pragma unroll
      for (int j = 0; j < 2; ++j)
#pragma unroll
        for (int r = 0; r < 8; ++r)
          Pw[(r + 8 * lh) * LDP + j * 16 + ln] = (_Float16)p[j][r];
      v16h aP = make_a_frag(Pw + ln * LDP, lane);

      // O += P (16x32) @ V (32x128)
#pragma unroll
      for (int t = 0; t < 8; ++t) {
        v16h bV = make_b_frag(Vt[buf] + (t * 16 + ln) * LDV, lane);
        o[t] = __builtin_amdgcn_wmma_f32_16x16x32_f16(
            false, aP, false, bV, (short)0, o[t], false, false);
      }
    }
  }

  // normalize, fuse sigmoid(gate), write f16 ctx
  float invl[8];
#pragma unroll
  for (int r = 0; r < 8; ++r) invl[r] = 1.f / lrow[r];
#pragma unroll
  for (int t = 0; t < 8; ++t)
#pragma unroll
    for (int r = 0; r < 8; ++r) {
      const int row = qs + r + 8 * lh, d = t * 16 + ln;
      const float g = QG[(size_t)row * (2 * NH * HD) + h * (2 * HD) + HD + d];
      const float val = o[t][r] * invl[r] * (1.f / (1.f + __expf(-g)));
      ctx[(size_t)row * (NH * HD) + h * HD + d] = (_Float16)val;
    }
}

// ---------------------------------------------------------------------------
extern "C" void kernel_launch(void* const* d_in, const int* in_sizes, int n_in,
                              void* d_out, int out_size, void* d_ws, size_t ws_size,
                              hipStream_t stream) {
  (void)in_sizes; (void)n_in; (void)out_size; (void)ws_size;
  const float* x    = (const float*)d_in[0];
  const float* fcos = (const float*)d_in[1];
  const float* fsin = (const float*)d_in[2];
  const float* wq   = (const float*)d_in[3];
  const float* wk   = (const float*)d_in[4];
  const float* wvw  = (const float*)d_in[5];
  const float* wo   = (const float*)d_in[6];
  const float* qnw  = (const float*)d_in[7];
  const float* knw  = (const float*)d_in[8];
  const float* anw  = (const float*)d_in[9];
  const float* fnw  = (const float*)d_in[10];
  const float* w1   = (const float*)d_in[11];
  const float* w2   = (const float*)d_in[12];
  const float* w3   = (const float*)d_in[13];
  float* out = (float*)d_out;
  char*  ws  = (char*)d_ws;

  // -------- workspace arena (phase 2 aliases dead phase-1 buffers) --------
  _Float16* xn   = (_Float16*)(ws + 0);          // 8 MB   (dead after qkv)
  float*    qg   = (float*)   (ws + 8388608);    // 33.5MB (q+gate; dead after attn)
  float*    xk   = (float*)   (ws + 41943040);   // 8.4MB
  float*    xv   = (float*)   (ws + 50331648);   // 8.4MB
  _Float16* qf   = (_Float16*)(ws + 58720256);   // 8.4MB
  _Float16* kf   = (_Float16*)(ws + 67108864);   // 4.2MB
  _Float16* vf   = (_Float16*)(ws + 71303168);   // 4.2MB
  _Float16* ctx  = (_Float16*)(ws + 75497472);   // 8.4MB (dead after wo)
  float*    hres = (float*)   (ws + 0);          // 16.8MB (aliases xn/qg, dead)
  _Float16* hn   = (_Float16*)(ws + 16777216);   // 8.4MB
  float*    h1   = (float*)   (ws + 25165824);   // 46.1MB
  _Float16* hmid = (_Float16*)(ws + 71303168);   // 23.1MB (aliases vf/ctx, dead)

  // 1) pre-attention rmsnorm -> f16 activations
  rmsnorm_f16<<<S_LEN, 256, 0, stream>>>(x, anw, xn, DIM);
  // 2) q/gate, k, v projections (WMMA)
  gemm_f16<0><<<dim3(32, 16), 256, 0, stream>>>(xn, wq,  qg, nullptr, nullptr, S_LEN, 2 * NH * HD, DIM);
  gemm_f16<0><<<dim3(8, 16),  256, 0, stream>>>(xn, wk,  xk, nullptr, nullptr, S_LEN, NKV * HD,    DIM);
  gemm_f16<0><<<dim3(8, 16),  256, 0, stream>>>(xn, wvw, xv, nullptr, nullptr, S_LEN, NKV * HD,    DIM);
  // 3) per-head rmsnorm + rope, pack to f16
  qk_norm_rope<<<dim3(S_LEN, NH),  128, 0, stream>>>(qg, 2 * NH * HD, 2 * HD, qnw, fcos, fsin, qf, NH * HD);
  qk_norm_rope<<<dim3(S_LEN, NKV), 128, 0, stream>>>(xk, NKV * HD,    HD,     knw, fcos, fsin, kf, NKV * HD);
  cvt_f32_f16<<<2048, 256, 0, stream>>>(xv, vf, S_LEN * NKV * HD);
  // 4) causal flash attention + sigmoid-gate fuse (WMMA)
  flash_attn<<<dim3(16, NH), 256, 0, stream>>>(qf, kf, vf, qg, ctx);
  // 5) output projection + residual: h = x + ctx @ wo^T
  gemm_f16<1><<<dim3(16, 16), 256, 0, stream>>>(ctx, wo, hres, nullptr, x, S_LEN, DIM, NH * HD);
  // 6) ffn rmsnorm -> f16
  rmsnorm_f16<<<S_LEN, 256, 0, stream>>>(hres, fnw, hn, DIM);
  // 7) swiglu: h1 = hn @ w1^T ; hmid = silu(h1) * (hn @ w3^T) in f16
  gemm_f16<0><<<dim3(44, 16), 256, 0, stream>>>(hn, w1, h1, nullptr, nullptr, S_LEN, HIDDEN, DIM);
  gemm_f16<2><<<dim3(44, 16), 256, 0, stream>>>(hn, w3, nullptr, hmid, h1,     S_LEN, HIDDEN, DIM);
  // 8) down projection + residual: out = h + hmid @ w2^T
  gemm_f16<1><<<dim3(16, 16), 256, 0, stream>>>(hmid, w2, out, nullptr, hres,  S_LEN, DIM, HIDDEN);
}
